// SpatialFeatureModeling_36034775614085
// MI455X (gfx1250) — matrix operand order repaired and checked by reading, not compiled, not run on hardware
//
#include <hip/hip_runtime.h>
#include <hip/hip_bf16.h>

typedef __attribute__((ext_vector_type(16))) _Float16 v16h;
typedef __attribute__((ext_vector_type(8)))  float    v8f;

#define D_HID 256

// ---------------- degree / dinv ----------------

__global__ void init_deg_kernel(float* __restrict__ deg, int n) {
  int i = blockIdx.x * blockDim.x + threadIdx.x;
  if (i < n) deg[i] = 1.0f;  // self loop
}

__global__ void degree_kernel(const long long* __restrict__ dst, float* __restrict__ deg,
                              long long E) {
  long long i = (long long)blockIdx.x * blockDim.x + threadIdx.x;
  long long stride = (long long)gridDim.x * blockDim.x;
  for (; i < E; i += stride) {
    unsafeAtomicAdd(&deg[dst[i]], 1.0f);
  }
}

__global__ void rsqrt_kernel(float* __restrict__ deg, int n) {
  int i = blockIdx.x * blockDim.x + threadIdx.x;
  if (i < n) deg[i] = rsqrtf(deg[i]);
}

// ---------------- GEMM: out[m][n] = dinv[m] * sum_k A[m][k] * W[k][n] ----------------
// One wave32 computes a 16x64 output strip: 4 accumulators share one A fragment
// per K-chunk (4x less A traffic + 4x fewer f32->f16 converts for A).
// v_wmma_f32_16x16x32_f16 lane layouts (ISA 7.12.2):
//   A (16x32 f16): lanes 0-15 row m, K = kc+{0..7,16..23}; lanes 16-31: kc+{8..15,24..31}
//   B (32x16 f16): lane col n = lane&15; lanes 0-15 K = kc+{0..7,16..23}; lanes 16-31: kc+{8..15,24..31}
//   C/D: VGPR r -> row 16*mtile + r + (lane<16?0:8), col = ntile*16 + (lane&15)

__global__ void gemm_wmma_scale_kernel(const float* __restrict__ A,
                                       const float* __restrict__ W,
                                       const float* __restrict__ dinv,
                                       float* __restrict__ out,
                                       int M, int K) {
  const int wavesPerBlock = blockDim.x >> 5;
  int wave = threadIdx.x >> 5;
  int lane = threadIdx.x & 31;
  int wv = blockIdx.x * wavesPerBlock + wave;
  int totalWaves = (M >> 4) * (D_HID >> 6);   // 16-row x 64-col strips
  if (wv >= totalWaves) return;

  int mtile = wv >> 2;            // D_HID/64 == 4 strips per row-block
  int nquad = wv & 3;
  int hl  = (lane < 16) ? 0 : 1;  // lane half
  int m   = (mtile << 4) + (lane & 15);
  int n0  = (nquad << 6) + (lane & 15);
  int ah  = hl ? 8 : 0;           // A K-offset for this half
  int bh  = hl ? 8 : 0;           // B K-offset for this half

  v8f acc0 = {}, acc1 = {}, acc2 = {}, acc3 = {};

  for (int kc = 0; kc < K; kc += 32) {
    const float* arow = A + (size_t)m * K + kc + ah;
    float4 a0 = *(const float4*)(arow + 0);
    float4 a1 = *(const float4*)(arow + 4);
    float4 a2 = *(const float4*)(arow + 16);
    float4 a3 = *(const float4*)(arow + 20);
    v16h av;
    av[0]  = (_Float16)a0.x; av[1]  = (_Float16)a0.y;
    av[2]  = (_Float16)a0.z; av[3]  = (_Float16)a0.w;
    av[4]  = (_Float16)a1.x; av[5]  = (_Float16)a1.y;
    av[6]  = (_Float16)a1.z; av[7]  = (_Float16)a1.w;
    av[8]  = (_Float16)a2.x; av[9]  = (_Float16)a2.y;
    av[10] = (_Float16)a2.z; av[11] = (_Float16)a2.w;
    av[12] = (_Float16)a3.x; av[13] = (_Float16)a3.y;
    av[14] = (_Float16)a3.z; av[15] = (_Float16)a3.w;

    const float* wbase = W + (size_t)(kc + bh) * D_HID + n0;
    v16h bv0, bv1, bv2, bv3;
#pragma unroll
    for (int j = 0; j < 16; ++j) {
      const float* wj = wbase + (size_t)(j + (j >= 8 ? 8 : 0)) * D_HID;
      bv0[j] = (_Float16)wj[0];
      bv1[j] = (_Float16)wj[16];
      bv2[j] = (_Float16)wj[32];
      bv3[j] = (_Float16)wj[48];
    }

    acc0 = __builtin_amdgcn_wmma_f32_16x16x32_f16(false, av, false, bv0, (short)0, acc0, false, false);
    acc1 = __builtin_amdgcn_wmma_f32_16x16x32_f16(false, av, false, bv1, (short)0, acc1, false, false);
    acc2 = __builtin_amdgcn_wmma_f32_16x16x32_f16(false, av, false, bv2, (short)0, acc2, false, false);
    acc3 = __builtin_amdgcn_wmma_f32_16x16x32_f16(false, av, false, bv3, (short)0, acc3, false, false);
  }

  int rbase = (mtile << 4) + (hl ? 8 : 0);
#pragma unroll
  for (int r = 0; r < 8; ++r) {
    int row = rbase + r;
    float di = dinv[row];
    float* orow = out + (size_t)row * D_HID + n0;
    orow[0]  = acc0[r] * di;
    orow[16] = acc1[r] * di;
    orow[32] = acc2[r] * di;
    orow[48] = acc3[r] * di;
  }
}

// ---------------- zero ----------------

__global__ void zero_kernel(float4* __restrict__ p, long long n4) {
  long long i = (long long)blockIdx.x * blockDim.x + threadIdx.x;
  long long stride = (long long)gridDim.x * blockDim.x;
  float4 z = {0.f, 0.f, 0.f, 0.f};
  for (; i < n4; i += stride) p[i] = z;
}

// ---------------- edge scatter: agg[dst] += hs[src] ----------------
// 64 threads per edge, float4 per thread (256 features).

__global__ void edge_scatter_kernel(const long long* __restrict__ src,
                                    const long long* __restrict__ dst,
                                    const float* __restrict__ hs,
                                    float* __restrict__ agg,
                                    long long E) {
  long long e = (long long)blockIdx.x * 4 + (threadIdx.x >> 6);
  if (e >= E) return;
  int t = (threadIdx.x & 63) << 2;
  long long s = src[e];
  long long d = dst[e];
  const float4 v = *(const float4*)(hs + s * D_HID + t);
  float* o = agg + d * D_HID + t;
  unsafeAtomicAdd(o + 0, v.x);
  unsafeAtomicAdd(o + 1, v.y);
  unsafeAtomicAdd(o + 2, v.z);
  unsafeAtomicAdd(o + 3, v.w);
}

// ---------------- finalize: out = dinv[row]*(agg + hs) + b[col]; optional relu ----------

__global__ void finalize_kernel(const float* __restrict__ agg,
                                const float* __restrict__ hs,
                                const float* __restrict__ dinv,
                                const float* __restrict__ bias,
                                float* __restrict__ out,
                                int N, int do_relu) {
  long long i4 = (long long)blockIdx.x * blockDim.x + threadIdx.x;
  long long total = (long long)N * (D_HID / 4);
  if (i4 >= total) return;
  long long base = i4 * 4;
  int row = (int)(base >> 8);
  int col = (int)(base & (D_HID - 1));
  float4 a = *(const float4*)(agg + base);
  float4 h = *(const float4*)(hs + base);
  float di = dinv[row];
  float4 r;
  r.x = di * (a.x + h.x) + bias[col + 0];
  r.y = di * (a.y + h.y) + bias[col + 1];
  r.z = di * (a.z + h.z) + bias[col + 2];
  r.w = di * (a.w + h.w) + bias[col + 3];
  if (do_relu) {
    r.x = fmaxf(r.x, 0.f); r.y = fmaxf(r.y, 0.f);
    r.z = fmaxf(r.z, 0.f); r.w = fmaxf(r.w, 0.f);
  }
  *(float4*)(out + base) = r;
}

// ---------------- launch ----------------

extern "C" void kernel_launch(void* const* d_in, const int* in_sizes, int n_in,
                              void* d_out, int out_size, void* d_ws, size_t ws_size,
                              hipStream_t stream) {
  const float*     x  = (const float*)d_in[0];
  const long long* ei = (const long long*)d_in[1];
  const float*     W1 = (const float*)d_in[2];
  const float*     b1 = (const float*)d_in[3];
  const float*     W2 = (const float*)d_in[4];
  const float*     b2 = (const float*)d_in[5];

  const int  D_IN = 128;
  const int  N    = in_sizes[0] / D_IN;           // 100000
  const long long E = (long long)in_sizes[1] / 2; // 3200000
  const long long* src = ei;
  const long long* dst = ei + E;

  float* dinv = (float*)d_ws;
  float* buf0 = dinv + N;                         // hs (prescaled GEMM out)
  float* buf1 = buf0 + (size_t)N * D_HID;         // aggregation buffer
  float* outp = (float*)d_out;

  const long long feat4 = (long long)N * (D_HID / 4);

  // degrees -> dinv
  init_deg_kernel<<<(N + 255) / 256, 256, 0, stream>>>(dinv, N);
  degree_kernel<<<2048, 256, 0, stream>>>(dst, dinv, E);
  rsqrt_kernel<<<(N + 255) / 256, 256, 0, stream>>>(dinv, N);

  int gemmWaves = (N / 16) * (D_HID / 64);
  int gemmGrid = (gemmWaves + 7) / 8;
  long long edgeGrid = (E + 3) / 4;
  int finGrid = (int)((feat4 + 255) / 256);

  // ---- layer 1: hs1 = dinv .* (x @ W1) ----
  gemm_wmma_scale_kernel<<<gemmGrid, 256, 0, stream>>>(x, W1, dinv, buf0, N, D_IN);
  zero_kernel<<<4096, 256, 0, stream>>>((float4*)buf1, feat4);
  edge_scatter_kernel<<<(int)edgeGrid, 256, 0, stream>>>(src, dst, buf0, buf1, E);
  finalize_kernel<<<finGrid, 256, 0, stream>>>(buf1, buf0, dinv, b1, buf1, N, /*relu=*/1);

  // ---- layer 2: hs2 = dinv .* (h1 @ W2) ----
  gemm_wmma_scale_kernel<<<gemmGrid, 256, 0, stream>>>(buf1, W2, dinv, buf0, N, D_HID);
  zero_kernel<<<4096, 256, 0, stream>>>((float4*)outp, feat4);
  edge_scatter_kernel<<<(int)edgeGrid, 256, 0, stream>>>(src, dst, buf0, outp, E);
  finalize_kernel<<<finGrid, 256, 0, stream>>>(outp, buf0, dinv, b2, outp, N, /*relu=*/0);
}
